// ShiftFuseZero_88064009437872
// MI455X (gfx1250) — matrix operand order repaired and checked by reading, not compiled, not run on hardware
//
#include <hip/hip_runtime.h>

typedef __attribute__((ext_vector_type(16))) _Float16 v16h;
typedef __attribute__((ext_vector_type(8)))  float    v8f;

#define B_  32
#define C_  96
#define T_  256
#define V_  25
#define N_  (T_*V_)      // 6400
#define KC_GCN 12        // K = 384 = 12*32
#define KC_TCN 3         // K = 96  = 3*32
#define OM_ 6            // M = 96 = 6*16
#define NT_ 400          // N = 6400 = 400*16

__device__ __forceinline__ float hswish_(float x){ return x * fminf(fmaxf(x + 3.f, 0.f), 6.f) * (1.f/6.f); }
__device__ __forceinline__ float sigm_(float x){ return 1.f/(1.f + expf(-x)); }

// ---------------------------------------------------------------------------
// Kernel 0: pack Wext[o, 0:384] = [W_gcn[k,o,c]/3 ; sigmoid(gate)*W_adyn[o,c]]
// into WMMA A-fragment layout: [om][kc][lane(32)][slot(16)] f16.
// Lane L: m = L%16, h = L/16; slot s -> k = kc*32 + h*16 + s
// ---------------------------------------------------------------------------
__global__ __launch_bounds__(256) void pack_gcn_A(const float* __restrict__ Wg,
                                                  const float* __restrict__ Wad,
                                                  const float* __restrict__ gate,
                                                  _Float16* __restrict__ Ap){
  int p = blockIdx.x * 256 + threadIdx.x;
  if (p >= OM_ * KC_GCN * 512) return;
  int slot = p & 15;
  int lane = (p >> 4) & 31;
  int kc   = (p >> 9) % KC_GCN;
  int om   = p / (KC_GCN * 512);
  int m = om*16 + (lane & 15);
  int h = lane >> 4;
  int k = kc*32 + h*16 + slot;
  float v;
  if (k < 288){ int part = k / 96, c = k % 96; v = Wg[(part*96 + m)*96 + c] * (1.f/3.f); }
  else        { int c = k - 288;               v = sigm_(gate[0]) * Wad[m*96 + c]; }
  Ap[p] = (_Float16)v;
}

// ---------------------------------------------------------------------------
// Kernel 1: per-batch dynamic adjacency A_dyn[b][25][25]
// ---------------------------------------------------------------------------
__global__ __launch_bounds__(256) void adyn_kernel(const float* __restrict__ x,
                                                   const int* __restrict__ intra,
                                                   const int* __restrict__ inter,
                                                   const float* __restrict__ Wp,
                                                   float* __restrict__ Adyn){
  __shared__ float xm[C_ * V_];
  __shared__ float em[24 * V_];
  __shared__ float inv[V_];
  int b = blockIdx.x, tid = threadIdx.x;
  for (int p = tid; p < C_ * V_; p += 256){
    int c = p / V_, v = p % V_;
    int src = (((c < 48) ? intra[v] : inter[v]) + c) % V_;
    const float* xp = x + (((size_t)b*C_ + c)*T_)*V_ + src;
    float s = 0.f;
    for (int t = 0; t < T_; ++t) s += xp[(size_t)t * V_];
    xm[p] = s * (1.f / T_);
  }
  __syncthreads();
  for (int p = tid; p < 24 * V_; p += 256){
    int e = p / V_, v = p % V_;
    float s = 0.f;
    for (int c = 0; c < C_; ++c) s += Wp[e*C_ + c] * xm[c*V_ + v];
    em[p] = s;
  }
  __syncthreads();
  if (tid < V_){
    float s = 0.f;
    for (int e = 0; e < 24; ++e){ float t = em[e*V_ + tid]; s += t*t; }
    inv[tid] = 1.f / (sqrtf(s) + 1e-12f);
  }
  __syncthreads();
  for (int p = tid; p < V_ * V_; p += 256){
    int v = p / V_, w = p % V_;
    float s = 0.f;
    for (int e = 0; e < 24; ++e) s += em[e*V_ + v] * em[e*V_ + w];
    Adyn[(size_t)b*V_*V_ + p] = s * inv[v] * inv[w];
  }
}

// ---------------------------------------------------------------------------
// Kernel 2: build Xhat (K_ext=384 rows) in WMMA B-fragment-friendly layout:
// Xh[b][kc][n][kk] f16, kk = k within 32-chunk.  Folds shift+gather.
// k = part*96+c for the 3 structural partitions, 288+c for A_dyn.
// ---------------------------------------------------------------------------
__global__ __launch_bounds__(256) void build_xhat(const float* __restrict__ x,
                                                  const int* __restrict__ intra,
                                                  const int* __restrict__ inter,
                                                  const float* __restrict__ Agcn,
                                                  const float* __restrict__ Adyn,
                                                  _Float16* __restrict__ Xh){
  __shared__ float Am[4][V_ * V_];
  int bid = blockIdx.x;          // = b*C_ + c
  int b = bid / C_, c = bid % C_;
  int t = threadIdx.x;           // 256 threads == T_
  for (int p = threadIdx.x; p < 3 * V_ * V_; p += 256) Am[p / (V_*V_)][p % (V_*V_)] = Agcn[p];
  for (int p = threadIdx.x; p < V_ * V_; p += 256)     Am[3][p] = Adyn[(size_t)b*V_*V_ + p];
  __syncthreads();
  const int* g = (c < 48) ? intra : inter;
  float xr[V_];
  const float* xp = x + (((size_t)b*C_ + c)*T_ + t)*V_;
  for (int v = 0; v < V_; ++v){ int src = (g[v] + c) % V_; xr[v] = xp[src]; }
  for (int part = 0; part < 4; ++part){
    int k  = (part < 3) ? (part*96 + c) : (288 + c);
    int kc = k >> 5, kk = k & 31;
    _Float16* dst = Xh + ((size_t)(b*KC_GCN + kc) * N_) * 32 + kk;
    for (int w = 0; w < V_; ++w){
      float s = 0.f;
      for (int v = 0; v < V_; ++v) s += xr[v] * Am[part][v*V_ + w];
      dst[(size_t)(t*V_ + w) * 32] = (_Float16)s;
    }
  }
}

// ---------------------------------------------------------------------------
// Kernel 3: main WMMA GEMM  y = Wext(96x384) @ Xhat(384x6400) per batch,
// epilogue: gcn BN + hswish, write f16 packed for the next GEMM's B operand.
// One wave (32 threads) per 16x16 output tile.
// ---------------------------------------------------------------------------
__global__ __launch_bounds__(32) void gemm_gcn(const _Float16* __restrict__ Ap,
                                               const _Float16* __restrict__ Xh,
                                               const float* __restrict__ gbn,
                                               _Float16* __restrict__ Xact){
  int lane = threadIdx.x;
  int nt = blockIdx.x, om = blockIdx.y, b = blockIdx.z;
  int nloc = nt*16 + (lane & 15);
  int h = lane >> 4;
  v8f acc = {0.f,0.f,0.f,0.f,0.f,0.f,0.f,0.f};
  const _Float16* abase = Ap + om * (KC_GCN * 512) + lane * 16;
  const _Float16* bbase = Xh + ((size_t)b * KC_GCN) * (size_t)N_ * 32 + (size_t)nloc * 32 + h * 16;
  for (int kc = 0; kc < KC_GCN; ++kc){
    if (kc + 1 < KC_GCN) __builtin_prefetch(bbase + (size_t)(kc + 1) * N_ * 32, 0, 1);
    v16h af = *(const v16h*)(abase + kc * 512);
    v16h bf = *(const v16h*)(bbase + (size_t)kc * N_ * 32);
    acc = __builtin_amdgcn_wmma_f32_16x16x32_f16(false, af, false, bf, (short)0, acc, false, false);
  }
  for (int r = 0; r < 8; ++r){
    int o = om*16 + r + 8*h;
    float gam = gbn[o], bet = gbn[96+o], rm = gbn[192+o], rv = gbn[288+o];
    float val = (acc[r] - rm) * (gam * rsqrtf(rv + 1e-5f)) + bet;
    val = hswish_(val);
    Xact[(((size_t)b*3 + (o >> 5)) * N_ + nloc) * 32 + (o & 31)] = (_Float16)val;
  }
}

// ---------------------------------------------------------------------------
// Kernel 4: SE (mean -> MLP -> sigmoid) and pack per-batch SE-scaled TCN
// weights [pa;pb;pc;pd] (96x96) into WMMA A-fragment layout.
// ---------------------------------------------------------------------------
__global__ __launch_bounds__(256) void se_pack(const _Float16* __restrict__ Xact,
                                               const float* __restrict__ w1, const float* __restrict__ b1,
                                               const float* __restrict__ w2, const float* __restrict__ b2,
                                               const float* __restrict__ pa, const float* __restrict__ pb,
                                               const float* __restrict__ pc, const float* __restrict__ pd,
                                               _Float16* __restrict__ Atcn){
  __shared__ float red[256];
  __shared__ float sC[96];
  __shared__ float r1[24];
  __shared__ float z[96];
  int b = blockIdx.x, tid = threadIdx.x;
  for (int o = 0; o < 96; ++o){
    const _Float16* p = Xact + (((size_t)b*3 + (o >> 5)) * N_) * 32 + (o & 31);
    float s = 0.f;
    for (int n = tid; n < N_; n += 256) s += (float)p[(size_t)n * 32];
    red[tid] = s; __syncthreads();
    for (int st = 128; st > 0; st >>= 1){ if (tid < st) red[tid] += red[tid + st]; __syncthreads(); }
    if (tid == 0) sC[o] = red[0] * (1.f / N_);
    __syncthreads();
  }
  if (tid < 24){
    float s = b1[tid];
    for (int c = 0; c < 96; ++c) s += w1[tid*96 + c] * sC[c];
    r1[tid] = fmaxf(s, 0.f);
  }
  __syncthreads();
  if (tid < 96){
    float s = b2[tid];
    for (int j = 0; j < 24; ++j) s += w2[tid*24 + j] * r1[j];
    z[tid] = sigm_(s);
  }
  __syncthreads();
  for (int p = tid; p < OM_ * KC_TCN * 512; p += 256){
    int slot = p & 15, lane = (p >> 4) & 31;
    int kc = (p >> 9) % KC_TCN, om = p / (KC_TCN * 512);
    int m = om*16 + (lane & 15), h = lane >> 4;
    int k = kc*32 + h*16 + slot;
    int br = m / 24, co = m % 24;
    const float* W = (br == 0) ? pa : (br == 1) ? pb : (br == 2) ? pc : pd;
    Atcn[(size_t)b * (OM_ * KC_TCN * 512) + p] = (_Float16)(W[co*96 + k] * z[k]);
  }
}

// ---------------------------------------------------------------------------
// Kernel 5: TCN pointwise WMMA GEMM  u = (SE-scaled W_cat) @ Xact,
// epilogue: per-branch BN (+hswish for branches a,b), f32 out.
// ---------------------------------------------------------------------------
__global__ __launch_bounds__(32) void gemm_tcn(const _Float16* __restrict__ Atcn,
                                               const _Float16* __restrict__ Xact,
                                               const float* __restrict__ pabn, const float* __restrict__ pbbn,
                                               const float* __restrict__ pcbn, const float* __restrict__ pdbn,
                                               float* __restrict__ U){
  int lane = threadIdx.x;
  int nt = blockIdx.x, om = blockIdx.y, b = blockIdx.z;
  int nloc = nt*16 + (lane & 15);
  int h = lane >> 4;
  v8f acc = {0.f,0.f,0.f,0.f,0.f,0.f,0.f,0.f};
  const _Float16* abase = Atcn + (size_t)b * (OM_ * KC_TCN * 512) + om * (KC_TCN * 512) + lane * 16;
  const _Float16* bbase = Xact + ((size_t)b * KC_TCN) * (size_t)N_ * 32 + (size_t)nloc * 32 + h * 16;
  for (int kc = 0; kc < KC_TCN; ++kc){
    v16h af = *(const v16h*)(abase + kc * 512);
    v16h bf = *(const v16h*)(bbase + (size_t)kc * N_ * 32);
    acc = __builtin_amdgcn_wmma_f32_16x16x32_f16(false, af, false, bf, (short)0, acc, false, false);
  }
  for (int r = 0; r < 8; ++r){
    int o = om*16 + r + 8*h;
    int br = o / 24, co = o % 24;
    const float* bn = (br == 0) ? pabn : (br == 1) ? pbbn : (br == 2) ? pcbn : pdbn;
    float val = (acc[r] - bn[48 + co]) * (bn[co] * rsqrtf(bn[72 + co] + 1e-5f)) + bn[24 + co];
    if (br < 2) val = hswish_(val);
    U[((size_t)b*96 + o) * N_ + nloc] = val;
  }
}

// ---------------------------------------------------------------------------
// Kernel 6: temporal convs (dil 1 & 2) + their BN, temporal maxpool,
// concat + identity residual + hswish -> output (f32).
// ---------------------------------------------------------------------------
__global__ __launch_bounds__(256) void finalize(const float* __restrict__ U,
                                                const float* __restrict__ x0,
                                                const float* __restrict__ taw, const float* __restrict__ tabn,
                                                const float* __restrict__ tbw, const float* __restrict__ tbbn,
                                                float* __restrict__ out){
  int idx = blockIdx.x * 256 + threadIdx.x;
  if (idx >= B_ * T_ * V_) return;
  int v = idx % V_;
  int t = (idx / V_) % T_;
  int b = idx / (T_ * V_);
  const float* Ub = U + (size_t)b * 96 * N_;
  int n = t * V_ + v;
  size_t base = (((size_t)b * 96) * T_ + t) * V_ + v;   // channel stride = N_
  for (int o = 0; o < 24; ++o){
    // branch a: kernel-3 temporal conv, dilation 1
    float aa = 0.f;
    for (int tap = 0; tap < 3; ++tap){
      int tt = t + tap - 1;
      if (tt < 0 || tt >= T_) continue;
      const float* ur = Ub + tt * V_ + v;
      for (int i = 0; i < 24; ++i) aa += taw[(o*24 + i)*3 + tap] * ur[(size_t)i * N_];
    }
    aa = (aa - tabn[48 + o]) * (tabn[o] * rsqrtf(tabn[72 + o] + 1e-5f)) + tabn[24 + o];
    // branch b: dilation 2
    float bb = 0.f;
    for (int tap = 0; tap < 3; ++tap){
      int tt = t + 2*tap - 2;
      if (tt < 0 || tt >= T_) continue;
      const float* ur = Ub + (size_t)24 * N_ + tt * V_ + v;
      for (int i = 0; i < 24; ++i) bb += tbw[(o*24 + i)*3 + tap] * ur[(size_t)i * N_];
    }
    bb = (bb - tbbn[48 + o]) * (tbbn[o] * rsqrtf(tbbn[72 + o] + 1e-5f)) + tbbn[24 + o];
    // branch c: temporal maxpool of BN'd values
    float cc = -3.0e38f;
    for (int tap = 0; tap < 3; ++tap){
      int tt = t + tap - 1;
      if (tt < 0 || tt >= T_) continue;
      cc = fmaxf(cc, Ub[(size_t)(48 + o) * N_ + tt * V_ + v]);
    }
    // branch d: identity
    float dd = Ub[(size_t)(72 + o) * N_ + n];
    float r;
    r = x0[base + (size_t)o * N_];        out[base + (size_t)o * N_]        = hswish_(aa + r);
    r = x0[base + (size_t)(24 + o) * N_]; out[base + (size_t)(24 + o) * N_] = hswish_(bb + r);
    r = x0[base + (size_t)(48 + o) * N_]; out[base + (size_t)(48 + o) * N_] = hswish_(cc + r);
    r = x0[base + (size_t)(72 + o) * N_]; out[base + (size_t)(72 + o) * N_] = hswish_(dd + r);
  }
}

// ---------------------------------------------------------------------------
extern "C" void kernel_launch(void* const* d_in, const int* in_sizes, int n_in,
                              void* d_out, int out_size, void* d_ws, size_t ws_size,
                              hipStream_t stream) {
  (void)in_sizes; (void)n_in; (void)out_size; (void)ws_size;
  const float* x    = (const float*)d_in[0];
  const int*   intra= (const int*)  d_in[1];
  const int*   inter= (const int*)  d_in[2];
  const float* Agcn = (const float*)d_in[3];
  const float* Wgcn = (const float*)d_in[4];
  const float* gbn  = (const float*)d_in[5];
  const float* Wp   = (const float*)d_in[6];
  const float* Wad  = (const float*)d_in[7];
  const float* gate = (const float*)d_in[8];
  const float* sw1  = (const float*)d_in[9];
  const float* sb1  = (const float*)d_in[10];
  const float* sw2  = (const float*)d_in[11];
  const float* sb2  = (const float*)d_in[12];
  const float* paw  = (const float*)d_in[13];
  const float* pabn = (const float*)d_in[14];
  const float* taw  = (const float*)d_in[15];
  const float* tabn = (const float*)d_in[16];
  const float* pbw  = (const float*)d_in[17];
  const float* pbbn = (const float*)d_in[18];
  const float* tbw  = (const float*)d_in[19];
  const float* tbbn = (const float*)d_in[20];
  const float* pcw  = (const float*)d_in[21];
  const float* pcbn = (const float*)d_in[22];
  const float* pdw  = (const float*)d_in[23];
  const float* pdbn = (const float*)d_in[24];

  char* ws = (char*)d_ws;
  size_t off = 0;
  auto carve = [&](size_t bytes) -> char* {
    char* p = ws + off;
    off = (off + bytes + 255) & ~(size_t)255;
    return p;
  };
  _Float16* ApG  = (_Float16*)carve((size_t)OM_ * KC_GCN * 512 * 2);            // packed Wext
  float*    Adyn = (float*)   carve((size_t)B_ * V_ * V_ * 4);                  // dynamic adjacency
  _Float16* Xh   = (_Float16*)carve((size_t)B_ * KC_GCN * N_ * 32 * 2);         // 157 MB
  _Float16* Xact = (_Float16*)carve((size_t)B_ * KC_TCN * N_ * 32 * 2);         // 39 MB
  _Float16* Atcn = (_Float16*)carve((size_t)B_ * OM_ * KC_TCN * 512 * 2);       // per-batch SE-scaled W
  float*    U    = (float*)   carve((size_t)B_ * 96 * N_ * 4);                  // 79 MB

  pack_gcn_A<<<(OM_*KC_GCN*512 + 255)/256, 256, 0, stream>>>(Wgcn, Wad, gate, ApG);
  adyn_kernel<<<B_, 256, 0, stream>>>(x, intra, inter, Wp, Adyn);
  build_xhat<<<B_ * C_, 256, 0, stream>>>(x, intra, inter, Agcn, Adyn, Xh);
  gemm_gcn<<<dim3(NT_, OM_, B_), 32, 0, stream>>>(ApG, Xh, gbn, Xact);
  se_pack<<<B_, 256, 0, stream>>>(Xact, sw1, sb1, sw2, sb2, paw, pbw, pcw, pdw, Atcn);
  gemm_tcn<<<dim3(NT_, OM_, B_), 32, 0, stream>>>(Atcn, Xact, pabn, pbbn, pcbn, pdbn, U);
  finalize<<<(B_*T_*V_ + 255)/256, 256, 0, stream>>>(U, x, taw, tabn, tbw, tbbn, (float*)d_out);
}
